// TransNeXtUNet_32109175504964
// MI455X (gfx1250) — matrix operand order repaired and checked by reading, not compiled, not run on hardware
//
#include <hip/hip_runtime.h>
#include <hip/hip_bf16.h>
#include <math.h>

typedef __bf16 bf16;
typedef __attribute__((ext_vector_type(16))) __bf16 v16bf;
typedef __attribute__((ext_vector_type(8)))  float  v8f;

#define BB 4
#define HH 64
#define WW 64
#define NN 4096
#define CC 256
#define NHD 8
#define HD 32
#define PL 64
#define LL 9

union Frag { v16bf v; unsigned u[8]; unsigned short s[16]; };

__device__ inline float wsum(float x) {
#pragma unroll
  for (int o = 16; o > 0; o >>= 1) x += __shfl_xor(x, o, 32);
  return x;
}
__device__ inline float wmaxr(float x) {
#pragma unroll
  for (int o = 16; o > 0; o >>= 1) x = fmaxf(x, __shfl_xor(x, o, 32));
  return x;
}

// ---------------- f32 -> bf16 convert (grid-stride) ----------------
__global__ void f2bf_k(const float* __restrict__ in, bf16* __restrict__ out, long n) {
  long i = (long)blockIdx.x * blockDim.x + threadIdx.x;
  long st = (long)gridDim.x * blockDim.x;
  for (; i < n; i += st) out[i] = (bf16)in[i];
}

// ---------------- generic bf16 WMMA GEMM (templated epilogue) ----------------
// D[z][m][n] = sum_k A[z][m][k] * W[n][k]   (B always weight layout, row-major [Nout][K])
// One wave computes a 64x32 tile: 4 M-subtiles x 2 N-subtiles = 8 WMMAs per K-step.
// EPI 0: store + bias; EPI 1: store gelu(val+bias); EPI 2: accumulate (+=)
template <int EPI>
__global__ __launch_bounds__(32)
void gemm_wmma(const bf16* __restrict__ A, long sAz, int lda,
               const bf16* __restrict__ Bm, long sBz, int ldb,
               float* __restrict__ D, int zdiv, long sDza, long sDzb, int ldd,
               const float* __restrict__ bias, int K) {
  const int lane = threadIdx.x;
  const int l15 = lane & 15, hi = lane >> 4;
  const int z = blockIdx.z;
  const bf16* Az = A + (long)z * sAz;
  const bf16* Bz = Bm + (long)z * sBz;
  float* Dz = D + (long)(z / zdiv) * sDza + (long)(z % zdiv) * sDzb;
  const int m0 = blockIdx.x * 64;
  const int n0 = blockIdx.y * 32;
  const int kA = hi * 16;  // A frag: lanes 16-31 cover K+16..31
  const int kB = hi * 16;  // B frag: lanes 16-31 cover K+16..31

  v8f acc[4][2];
  const v8f vzero = {0.f, 0.f, 0.f, 0.f, 0.f, 0.f, 0.f, 0.f};
#pragma unroll
  for (int s = 0; s < 4; ++s) {
    acc[s][0] = vzero;
    acc[s][1] = vzero;
  }

  for (int k0 = 0; k0 < K; k0 += 32) {
    Frag bf0, bf1;
    {
      const bf16* bp0 = Bz + (long)(n0 + l15) * ldb + k0 + kB;
      const bf16* bp1 = Bz + (long)(n0 + 16 + l15) * ldb + k0 + kB;
      __builtin_prefetch(bp0 + 32, 0, 1);
      __builtin_prefetch(bp1 + 32, 0, 1);
#pragma unroll
      for (int v = 0; v < 8; ++v) {
        bf0.u[v] = *(const unsigned*)(bp0 + 2 * v);
        bf1.u[v] = *(const unsigned*)(bp1 + 2 * v);
      }
    }
#pragma unroll
    for (int s = 0; s < 4; ++s) {
      Frag af;
      const bf16* ap = Az + (long)(m0 + s * 16 + l15) * lda + k0;
      __builtin_prefetch(ap + 32, 0, 1);
#pragma unroll
      for (int v = 0; v < 8; ++v) {
        int koff = kA + v * 2;
        af.u[v] = *(const unsigned*)(ap + koff);
      }
      acc[s][0] = __builtin_amdgcn_wmma_f32_16x16x32_bf16(
          false, af.v, false, bf0.v, (short)0, acc[s][0], false, false);
      acc[s][1] = __builtin_amdgcn_wmma_f32_16x16x32_bf16(
          false, af.v, false, bf1.v, (short)0, acc[s][1], false, false);
    }
  }

#pragma unroll
  for (int t = 0; t < 2; ++t) {
    const int col = n0 + t * 16 + l15;
    float bcol = 0.0f;
    if (EPI != 2) bcol = bias[col];
#pragma unroll
    for (int s = 0; s < 4; ++s) {
#pragma unroll
      for (int r = 0; r < 8; ++r) {
        int m = m0 + s * 16 + r + hi * 8;
        float val = acc[s][t][r];
        float* dp = Dz + (long)m * ldd + col;
        if (EPI == 0) {
          *dp = val + bcol;
        } else if (EPI == 1) {
          float xx = val + bcol;
          *dp = 0.5f * xx * (1.0f + erff(xx * 0.70710678118654752f));
        } else {
          *dp += val;
        }
      }
    }
  }
}

// ---------------- attn_pool = q_scaled @ k_pool_norm^T + cpb gather ----------------
// A: [z][N][32] bf16, B: [z][64][32] bf16 (weight layout), D: [z][N][64] f32
// 64x32 tile per wave, single K-step (K = HD = 32).
__global__ __launch_bounds__(32)
void attn_pool_wmma(const bf16* __restrict__ Q, const bf16* __restrict__ Kp,
                    const float* __restrict__ cpb, const int* __restrict__ rpi,
                    float* __restrict__ AP) {
  const int lane = threadIdx.x;
  const int l15 = lane & 15, hi = lane >> 4;
  const int z = blockIdx.z, h = z & 7;
  const bf16* Az = Q + (long)z * NN * HD;
  const bf16* Bz = Kp + (long)z * PL * HD;
  float* Dz = AP + (long)z * NN * PL;
  const int m0 = blockIdx.x * 64;
  const int n0 = blockIdx.y * 32;

  v8f acc[4][2];
  const v8f vzero = {0.f, 0.f, 0.f, 0.f, 0.f, 0.f, 0.f, 0.f};
#pragma unroll
  for (int s = 0; s < 4; ++s) {
    acc[s][0] = vzero;
    acc[s][1] = vzero;
  }

  Frag bf0, bf1;
  const bf16* bp0 = Bz + (long)(n0 + l15) * HD + hi * 16;
  const bf16* bp1 = Bz + (long)(n0 + 16 + l15) * HD + hi * 16;
#pragma unroll
  for (int v = 0; v < 8; ++v) {
    bf0.u[v] = *(const unsigned*)(bp0 + 2 * v);
    bf1.u[v] = *(const unsigned*)(bp1 + 2 * v);
  }
#pragma unroll
  for (int s = 0; s < 4; ++s) {
    Frag af;
    const bf16* ap = Az + (long)(m0 + s * 16 + l15) * HD;
#pragma unroll
    for (int v = 0; v < 8; ++v) {
      int koff = hi * 16 + v * 2;
      af.u[v] = *(const unsigned*)(ap + koff);
    }
    acc[s][0] = __builtin_amdgcn_wmma_f32_16x16x32_bf16(
        false, af.v, false, bf0.v, (short)0, acc[s][0], false, false);
    acc[s][1] = __builtin_amdgcn_wmma_f32_16x16x32_bf16(
        false, af.v, false, bf1.v, (short)0, acc[s][1], false, false);
  }

#pragma unroll
  for (int t = 0; t < 2; ++t) {
    const int p = n0 + t * 16 + l15;
#pragma unroll
    for (int s = 0; s < 4; ++s) {
#pragma unroll
      for (int r = 0; r < 8; ++r) {
        int m = m0 + s * 16 + r + hi * 8;
        float bias = cpb[(long)rpi[(long)m * PL + p] * NHD + h];
        Dz[(long)m * PL + p] = acc[s][t][r] + bias;
      }
    }
  }
}

// ---------------- q normalize + scale (one wave per (b,n,h), lane=d) ----------------
__global__ __launch_bounds__(256)
void q_process_k(const float* __restrict__ q, const float* __restrict__ qe,
                 const float* __restrict__ temp, const float* __restrict__ sls,
                 float* __restrict__ qn_out, bf16* __restrict__ qs_out) {
  int lane = threadIdx.x & 31;
  long wave = ((long)blockIdx.x * blockDim.x + threadIdx.x) >> 5;  // (b*N+n)*8+h
  int h = (int)(wave & 7);
  long bn = wave >> 3;
  int n = (int)(bn % NN);
  int b = (int)(bn / NN);
  float v = q[bn * CC + h * HD + lane];
  float nrm = fmaxf(sqrtf(wsum(v * v)), 1e-12f);
  float qn = v / nrm;
  long zo = (((long)(b * NHD + h) * NN + n)) * HD + lane;
  qn_out[zo] = qn;
  float sp = log1pf(expf(temp[h]));
  qs_out[zo] = (bf16)((qn + qe[h * HD + lane]) * sp * sls[n]);
}

// ---------------- normalize k_loc in place inside kv (first C channels) ----------------
__global__ __launch_bounds__(256)
void kloc_norm_k(float* __restrict__ kv) {
  int lane = threadIdx.x & 31;
  long wave = ((long)blockIdx.x * blockDim.x + threadIdx.x) >> 5;  // (b*N+n)*8+h
  int h = (int)(wave & 7);
  long bn = wave >> 3;
  long idx = bn * (2 * CC) + h * HD + lane;
  float v = kv[idx];
  float nrm = fmaxf(sqrtf(wsum(v * v)), 1e-12f);
  kv[idx] = v / nrm;
}

// -------- pooled kv: normalize k_pool (bf16 [z][p][d]), transpose v_pool (bf16 [z][d][p]) --------
__global__ __launch_bounds__(256)
void kvp_process_k(const float* __restrict__ kvp, bf16* __restrict__ kpn,
                   bf16* __restrict__ vpT) {
  int lane = threadIdx.x & 31;
  long wave = ((long)blockIdx.x * blockDim.x + threadIdx.x) >> 5;  // (b*64+p)*8+h
  int h = (int)(wave & 7);
  long bp = wave >> 3;
  int p = (int)(bp & 63);
  int b = (int)(bp >> 6);
  float kv = kvp[bp * (2 * CC) + h * HD + lane];
  float nrm = fmaxf(sqrtf(wsum(kv * kv)), 1e-12f);
  int zbh = b * NHD + h;
  kpn[((long)zbh * PL + p) * HD + lane] = (bf16)(kv / nrm);
  float vv = kvp[bp * (2 * CC) + CC + h * HD + lane];
  vpT[((long)zbh * HD + lane) * PL + p] = (bf16)vv;  // transposed: [d][p]
}

// ---------------- 8x8 avg-pool of GELU(sr) + LayerNorm over C ----------------
__global__ __launch_bounds__(256)
void pool_ln_k(const float* __restrict__ sr, const float* __restrict__ g,
               const float* __restrict__ bt, float* __restrict__ out,
               bf16* __restrict__ outb) {
  __shared__ float red[CC];
  int bp = blockIdx.x;
  int b = bp >> 6, p = bp & 63;
  int c = threadIdx.x;
  int ph = p >> 3, pw = p & 7;
  float s = 0.f;
#pragma unroll
  for (int r = 0; r < 8; ++r)
#pragma unroll
    for (int t = 0; t < 8; ++t) {
      int n = (ph * 8 + r) * WW + (pw * 8 + t);
      s += sr[((long)b * NN + n) * CC + c];
    }
  float v = s * (1.0f / 64.0f);
  red[c] = v;
  __syncthreads();
  for (int st = 128; st > 0; st >>= 1) {
    if (c < st) red[c] += red[c + st];
    __syncthreads();
  }
  float mean = red[0] * (1.0f / CC);
  __syncthreads();
  float dv = v - mean;
  red[c] = dv * dv;
  __syncthreads();
  for (int st = 128; st > 0; st >>= 1) {
    if (c < st) red[c] += red[c + st];
    __syncthreads();
  }
  float var = red[0] * (1.0f / CC);
  float o = dv * rsqrtf(var + 1e-5f) * g[c] + bt[c];
  long idx = (long)bp * CC + c;
  out[idx] = o;
  outb[idx] = (bf16)o;
}

// ---------------- CPB MLP: cpb[t][o] = relu(table@W1^T+b1)@W2^T+b2 ----------------
__global__ void cpb_k(const float* __restrict__ table, const float* __restrict__ w1,
                      const float* __restrict__ b1, const float* __restrict__ w2,
                      const float* __restrict__ b2, float* __restrict__ cpb, int T) {
  int tid = blockIdx.x * blockDim.x + threadIdx.x;
  if (tid >= T * NHD) return;
  int t = tid >> 3, o = tid & 7;
  float x0 = table[t * 2], x1 = table[t * 2 + 1];
  float acc = b2[o];
  for (int j = 0; j < 512; ++j) {
    float hj = fmaxf(x0 * w1[j * 2] + x1 * w1[j * 2 + 1] + b1[j], 0.f);
    acc += hj * w2[o * 512 + j];
  }
  cpb[t * NHD + o] = acc;
}

// ---------------- local attention logits (3x3 window, wave per (b,n,h)) ----------------
__global__ __launch_bounds__(256)
void attn_local_k(const bf16* __restrict__ qs, const float* __restrict__ kv,
                  const float* __restrict__ rbl, float* __restrict__ AL) {
  int lane = threadIdx.x & 31;
  long wave = ((long)blockIdx.x * blockDim.x + threadIdx.x) >> 5;  // (b*N+n)*8+h
  int h = (int)(wave & 7);
  long bn = wave >> 3;
  int n = (int)(bn % NN);
  int b = (int)(bn / NN);
  float qd = (float)qs[(((long)(b * NHD + h) * NN + n)) * HD + lane];
  int pi = n / WW, pj = n % WW;
#pragma unroll
  for (int l = 0; l < LL; ++l) {
    int ni = pi + l / 3 - 1, nj = pj + l % 3 - 1;
    bool valid = (ni >= 0) && (ni < HH) && (nj >= 0) && (nj < WW);
    float kl = valid ? kv[((long)b * NN + ni * WW + nj) * (2 * CC) + h * HD + lane] : 0.f;
    float s = wsum(qd * kl);
    if (lane == 0)
      AL[((long)(b * NHD + h) * NN + n) * LL + l] =
          valid ? (s + rbl[h * LL + l]) : -INFINITY;
  }
}

// ---------------- softmax over 9 local + 64 pool (wave per row) ----------------
__global__ __launch_bounds__(256)
void softmax_k(const float* __restrict__ AL, const float* __restrict__ AP,
               float* __restrict__ aloc, bf16* __restrict__ apool) {
  int lane = threadIdx.x & 31;
  long row = ((long)blockIdx.x * blockDim.x + threadIdx.x) >> 5;  // z*N+n
  float v0 = (lane < LL) ? AL[row * LL + lane] : AP[row * PL + (lane - LL)];
  float v1 = AP[row * PL + (lane + 32 - LL)];
  float v2 = (lane < LL) ? AP[row * PL + (lane + 64 - LL)] : -INFINITY;
  float m = wmaxr(fmaxf(v0, fmaxf(v1, v2)));
  float e0 = expf(v0 - m), e1 = expf(v1 - m);
  float e2 = (lane < LL) ? expf(v2 - m) : 0.f;
  float inv = 1.0f / wsum(e0 + e1 + e2);
  if (lane < LL) aloc[row * LL + lane] = e0 * inv;
  else apool[row * PL + (lane - LL)] = (bf16)(e0 * inv);
  apool[row * PL + (lane + 32 - LL)] = (bf16)(e1 * inv);
  if (lane < LL) apool[row * PL + (lane + 64 - LL)] = (bf16)(e2 * inv);
}

// ---------------- learnable tokens + local value aggregation ----------------
__global__ __launch_bounds__(256)
void xlocal_k(const float* __restrict__ qn, const float* __restrict__ kv,
              const float* __restrict__ ltok, const float* __restrict__ lbias,
              const float* __restrict__ aloc, float* __restrict__ xout) {
  int lane = threadIdx.x & 31;
  long wave = ((long)blockIdx.x * blockDim.x + threadIdx.x) >> 5;  // (b*N+n)*8+h
  int h = (int)(wave & 7);
  long bn = wave >> 3;
  int n = (int)(bn % NN);
  int b = (int)(bn / NN);
  long zrow = ((long)(b * NHD + h) * NN + n);
  float qv = qn[zrow * HD + lane];
  int pi = n / WW, pj = n % WW;
  float acc = 0.f;
#pragma unroll
  for (int l = 0; l < LL; ++l) {
    float tok = ltok[(h * HD + lane) * LL + l];
    float ltl = wsum(qv * tok);
    ltl += lbias[h * LL + l] + aloc[zrow * LL + l];
    int ni = pi + l / 3 - 1, nj = pj + l % 3 - 1;
    bool valid = (ni >= 0) && (ni < HH) && (nj >= 0) && (nj < WW);
    float v = valid ? kv[((long)b * NN + ni * WW + nj) * (2 * CC) + CC + h * HD + lane] : 0.f;
    acc += ltl * v;
  }
  xout[bn * CC + h * HD + lane] = acc;
}

// =========================== host launcher ===========================
extern "C" void kernel_launch(void* const* d_in, const int* in_sizes, int n_in,
                              void* d_out, int out_size, void* d_ws, size_t ws_size,
                              hipStream_t stream) {
  const float* x      = (const float*)d_in[0];
  const float* table  = (const float*)d_in[1];
  const float* sls    = (const float*)d_in[2];
  const float* q_w    = (const float*)d_in[3];
  const float* q_b    = (const float*)d_in[4];
  const float* kv_w   = (const float*)d_in[5];
  const float* kv_b   = (const float*)d_in[6];
  const float* temp   = (const float*)d_in[7];
  const float* qe     = (const float*)d_in[8];
  const float* sr_w   = (const float*)d_in[9];
  const float* sr_b   = (const float*)d_in[10];
  const float* norm_g = (const float*)d_in[11];
  const float* norm_b = (const float*)d_in[12];
  const float* w1     = (const float*)d_in[13];
  const float* b1     = (const float*)d_in[14];
  const float* w2     = (const float*)d_in[15];
  const float* b2     = (const float*)d_in[16];
  const float* rbl    = (const float*)d_in[17];
  const float* ltok   = (const float*)d_in[18];
  const float* lbias  = (const float*)d_in[19];
  const float* proj_w = (const float*)d_in[20];
  const float* proj_b = (const float*)d_in[21];
  const int*   rpi    = (const int*)d_in[22];
  const int T = in_sizes[1] / 2;

  // workspace bump allocator
  char* ws = (char*)d_ws;
  size_t off = 0;
  auto alloc = [&](size_t bytes) -> void* {
    void* p = ws + off;
    off = (off + bytes + 255) & ~(size_t)255;
    return p;
  };
  const long M = (long)BB * NN;  // 16384
  bf16* xb    = (bf16*)alloc(M * CC * 2);
  bf16* wq    = (bf16*)alloc(CC * CC * 2);
  bf16* wkv   = (bf16*)alloc(2 * CC * CC * 2);
  bf16* wsr   = (bf16*)alloc(CC * CC * 2);
  bf16* wproj = (bf16*)alloc(CC * CC * 2);
  float* qf   = (float*)alloc(M * CC * 4);
  float* kvf  = (float*)alloc(M * 2 * CC * 4);
  float* srf  = (float*)alloc(M * CC * 4);
  float* xpl  = (float*)alloc((long)BB * PL * CC * 4);
  bf16*  xplb = (bf16*)alloc((long)BB * PL * CC * 2);
  float* kvpf = (float*)alloc((long)BB * PL * 2 * CC * 4);
  float* qn   = (float*)alloc(M * CC * 4);
  bf16*  qs   = (bf16*)alloc(M * CC * 2);
  bf16*  kpn  = (bf16*)alloc((long)BB * NHD * PL * HD * 2);
  bf16*  vpT  = (bf16*)alloc((long)BB * NHD * HD * PL * 2);
  float* cpb  = (float*)alloc((long)T * NHD * 4);
  float* AL   = (float*)alloc((long)BB * NHD * NN * LL * 4);
  float* AP   = (float*)alloc((long)BB * NHD * NN * PL * 4);
  float* aloc = (float*)alloc((long)BB * NHD * NN * LL * 4);
  bf16*  apool= (bf16*)alloc((long)BB * NHD * NN * PL * 2);
  float* xout = (float*)alloc(M * CC * 4);
  bf16*  xob  = (bf16*)alloc(M * CC * 2);

  // 1) bf16 conversions
  f2bf_k<<<4096, 256, 0, stream>>>(x, xb, M * CC);
  f2bf_k<<<256, 256, 0, stream>>>(q_w, wq, CC * CC);
  f2bf_k<<<512, 256, 0, stream>>>(kv_w, wkv, 2 * CC * CC);
  f2bf_k<<<256, 256, 0, stream>>>(sr_w, wsr, CC * CC);
  f2bf_k<<<256, 256, 0, stream>>>(proj_w, wproj, CC * CC);

  // 2) q = x @ q_w^T + q_b
  gemm_wmma<0><<<dim3(M / 64, CC / 32, 1), 32, 0, stream>>>(
      xb, 0, CC, wq, 0, CC, qf, 1, 0, 0, CC, q_b, CC);
  // 3) kv = x @ kv_w^T + kv_b
  gemm_wmma<0><<<dim3(M / 64, 2 * CC / 32, 1), 32, 0, stream>>>(
      xb, 0, CC, wkv, 0, CC, kvf, 1, 0, 0, 2 * CC, kv_b, CC);
  // 4) sr = gelu(x @ sr_w^T + sr_b)
  gemm_wmma<1><<<dim3(M / 64, CC / 32, 1), 32, 0, stream>>>(
      xb, 0, CC, wsr, 0, CC, srf, 1, 0, 0, CC, sr_b, CC);
  // 5) 8x8 pool + LN
  pool_ln_k<<<BB * PL, CC, 0, stream>>>(srf, norm_g, norm_b, xpl, xplb);
  // 6) kv_p = x_pool @ kv_w^T + kv_b   (M = 256)
  gemm_wmma<0><<<dim3(BB * PL / 64, 2 * CC / 32, 1), 32, 0, stream>>>(
      xplb, 0, CC, wkv, 0, CC, kvpf, 1, 0, 0, 2 * CC, kv_b, CC);

  // 7) per-head normalizations / scaling
  q_process_k<<<(int)(M * NHD * 32 / 256), 256, 0, stream>>>(qf, qe, temp, sls, qn, qs);
  kloc_norm_k<<<(int)(M * NHD * 32 / 256), 256, 0, stream>>>(kvf);
  kvp_process_k<<<(int)((long)BB * PL * NHD * 32 / 256), 256, 0, stream>>>(kvpf, kpn, vpT);

  // 8) CPB relative-position MLP
  cpb_k<<<(T * NHD + 255) / 256, 256, 0, stream>>>(table, w1, b1, w2, b2, cpb, T);

  // 9) local attention logits
  attn_local_k<<<(int)(M * NHD * 32 / 256), 256, 0, stream>>>(qs, kvf, rbl, AL);
  // 10) pooled attention logits (WMMA, fused CPB bias gather)
  attn_pool_wmma<<<dim3(NN / 64, PL / 32, BB * NHD), 32, 0, stream>>>(qs, kpn, cpb, rpi, AP);
  // 11) softmax over [9 local | 64 pool]
  softmax_k<<<(int)((long)BB * NHD * NN * 32 / 256), 256, 0, stream>>>(AL, AP, aloc, apool);
  // 12) learnable tokens + local aggregation -> xout
  xlocal_k<<<(int)(M * NHD * 32 / 256), 256, 0, stream>>>(qn, kvf, ltok, lbias, aloc, xout);
  // 13) xout += a_pool @ v_pool   (B = vpT, weight layout [d][p]; per (b,h) strided D)
  gemm_wmma<2><<<dim3(NN / 64, HD / 32, BB * NHD), 32, 0, stream>>>(
      apool, (long)NN * PL, PL, vpT, (long)HD * PL, PL, xout, NHD,
      (long)NN * CC, HD, CC, nullptr, PL);
  // 14) out = xout @ proj_w^T + proj_b
  f2bf_k<<<4096, 256, 0, stream>>>(xout, xob, M * CC);
  gemm_wmma<0><<<dim3(M / 64, CC / 32, 1), 32, 0, stream>>>(
      xob, 0, CC, wproj, 0, CC, (float*)d_out, 1, 0, 0, CC, proj_b, CC);
}
